// MAMBA_35373350650147
// MI455X (gfx1250) — compile-verified, matrix operand order checked by reading, not run on hardware
//
#include <hip/hip_runtime.h>
#include <hip/hip_bf16.h>
#include <math.h>
#include <stdint.h>

// ---------------- model dims (compile-time) ----------------
#define BB      2
#define LL      1024
#define DD      1024
#define EE      2048          // d_inner
#define NLAYER  4
#define DSTATE  16
#define DCONV   4
#define DTRANK  64
#define XDB_LD  96            // dt_rank + 2*d_state (source layout)
#define XDB_PAD 128           // padded to a multiple of GEMM BN
#define MROWS   (BB * LL)     // 2048
#define EPS     1e-5f

// ---------------- WMMA / TDM vector types (gfx1250, wave32) ----------------
typedef __attribute__((ext_vector_type(16))) __bf16 v16bf;
typedef __attribute__((ext_vector_type(8)))  float  v8f;
typedef __attribute__((ext_vector_type(4)))  unsigned int v4u;
typedef __attribute__((ext_vector_type(4)))  int v4i;
typedef __attribute__((ext_vector_type(8)))  int v8i;

#if __has_builtin(__builtin_amdgcn_tensor_load_to_lds) && \
    __has_builtin(__builtin_amdgcn_s_wait_tensorcnt)
#define USE_TDM 1
#else
#define USE_TDM 0
#endif

// ---------------- fp32 -> bf16 helpers ----------------
__device__ __forceinline__ unsigned f2bf1(float f) {
    unsigned u = __builtin_bit_cast(unsigned, f);
    u += 0x7fffu + ((u >> 16) & 1u);      // round-to-nearest-even
    return u >> 16;
}
__device__ __forceinline__ unsigned pack2bf(float lo, float hi) {
#if __has_builtin(__builtin_amdgcn_cvt_pk_bf16_f32)
    typedef __attribute__((ext_vector_type(2))) __bf16 v2bf;
    v2bf r = __builtin_amdgcn_cvt_pk_bf16_f32(lo, hi);
    return __builtin_bit_cast(unsigned, r);
#else
    unsigned ulo = __builtin_bit_cast(unsigned, lo);
    unsigned uhi = __builtin_bit_cast(unsigned, hi);
    ulo += 0x7fffu + ((ulo >> 16) & 1u);
    uhi += 0x7fffu + ((uhi >> 16) & 1u);
    return (ulo >> 16) | (uhi & 0xffff0000u);
#endif
}

// =====================================================================
// Bulk fp32 -> bf16 conversion (n4 = quads).
// =====================================================================
__global__ __launch_bounds__(256)
void cvt_bf16_kernel(const float* __restrict__ src, unsigned short* __restrict__ dst,
                     long long n4) {
    const long long i = (long long)blockIdx.x * 256 + threadIdx.x;
    if (i >= n4) return;
    float4 v = ((const float4*)src)[i];
    uint2 o; o.x = pack2bf(v.x, v.y); o.y = pack2bf(v.z, v.w);
    ((uint2*)dst)[i] = o;
}

// =====================================================================
// Zero-pad + convert x_proj weights (NLAYER x 96 x E) -> bf16 (NLAYER x 128 x E).
// =====================================================================
__global__ __launch_bounds__(256)
void pad_xpw_bf16_kernel(const float* __restrict__ xpw, unsigned short* __restrict__ xpad) {
    const int i = blockIdx.x * 256 + threadIdx.x;       // over NLAYER*128*EE
    if (i >= NLAYER * XDB_PAD * EE) return;
    const int layer = i / (XDB_PAD * EE);
    const int rem   = i % (XDB_PAD * EE);
    const int row   = rem / EE, col = rem % EE;
    xpad[i] = (row < XDB_LD)
        ? (unsigned short)f2bf1(xpw[((size_t)layer * XDB_LD + row) * EE + col])
        : (unsigned short)0;
}

// =====================================================================
// Fused (residual add) + RMSNorm.  One block per row of D=1024 floats.
// Emits fp32 (final norm) and/or bf16 (GEMM feed).
// =====================================================================
__global__ __launch_bounds__(256)
void addnorm_kernel(const float* __restrict__ h, const float* __restrict__ res_in,
                    float* __restrict__ res_out, const float* __restrict__ w,
                    float* __restrict__ out_f32, unsigned short* __restrict__ out_bf,
                    int addres) {
    const int row = blockIdx.x;
    const int t   = threadIdx.x;
    const size_t base = (size_t)row * DD;

    float4 hv = *(const float4*)(h + base + t * 4);
    if (addres) {
        float4 rv = *(const float4*)(res_in + base + t * 4);
        hv.x += rv.x; hv.y += rv.y; hv.z += rv.z; hv.w += rv.w;
    }
    *(float4*)(res_out + base + t * 4) = hv;

    float ss = hv.x * hv.x + hv.y * hv.y + hv.z * hv.z + hv.w * hv.w;
    #pragma unroll
    for (int m = 16; m > 0; m >>= 1) ss += __shfl_xor(ss, m, 32);   // wave32

    __shared__ float part[8];
    __shared__ float scale_s;
    const int lane = t & 31, wave = t >> 5;
    if (lane == 0) part[wave] = ss;
    __syncthreads();
    if (t == 0) {
        float tot = 0.f;
        #pragma unroll
        for (int i = 0; i < 8; ++i) tot += part[i];
        scale_s = rsqrtf(tot / (float)DD + EPS);
    }
    __syncthreads();
    const float sc = scale_s;

    float4 wv = *(const float4*)(w + t * 4);
    float4 ov;
    ov.x = hv.x * sc * wv.x; ov.y = hv.y * sc * wv.y;
    ov.z = hv.z * sc * wv.z; ov.w = hv.w * sc * wv.w;
    if (out_f32) *(float4*)(out_f32 + base + t * 4) = ov;
    if (out_bf) {
        uint2 o; o.x = pack2bf(ov.x, ov.y); o.y = pack2bf(ov.z, ov.w);
        *(uint2*)(out_bf + base + t * 4) = o;
    }
}

// =====================================================================
// TDM descriptor build + issue (2D tile of bf16, hardware LDS row padding:
// 16 dwords data + 4 dwords pad -> 40-element padded stride).
// Fields per CDNA5 ISA.
// =====================================================================
#if USE_TDM
__device__ __forceinline__ void tdm_load_tile_2d(unsigned lds_byte_addr,
                                                 const unsigned short* g,
                                                 int tensor_d0, int tensor_d1,
                                                 int stride_elems,
                                                 int tile0, int tile1) {
    unsigned long long ga = (unsigned long long)(uintptr_t)g;
    v4u g0;
    g0[0] = 1u;                                               // count=1 (user D#)
    g0[1] = lds_byte_addr;                                    // lds_addr
    g0[2] = (unsigned)(ga & 0xffffffffull);                   // global_addr[31:0]
    g0[3] = (unsigned)((ga >> 32) & 0x01ffffffull)            // global_addr[56:32]
          | (2u << 30);                                       // type = 2 (image)
    const unsigned w0 = (1u << 16)   // data_size = 2 bytes
                      | (1u << 20)   // pad_enable
                      | (3u << 22)   // pad_interval: 16 dwords
                      | (3u << 25);  // pad_amount:   4 dwords
    const unsigned w1 = ((unsigned)tensor_d0 & 0xffffu) << 16;
    const unsigned w2 = (((unsigned)tensor_d0 >> 16) & 0xffffu)
                      | (((unsigned)tensor_d1 & 0xffffu) << 16);
    const unsigned w3 = (((unsigned)tensor_d1 >> 16) & 0xffffu)
                      | ((unsigned)tile0 << 16);
    const unsigned w4 = (unsigned)tile1 & 0xffffu;            // tile_dim1 (tile_dim2=0)
    v8i g1;
    g1[0] = (int)w0; g1[1] = (int)w1; g1[2] = (int)w2; g1[3] = (int)w3;
    g1[4] = (int)w4; g1[5] = stride_elems; g1[6] = 0; g1[7] = 0;
    v4i z = {0, 0, 0, 0};
#if defined(__clang_major__) && __clang_major__ >= 23
    v8i z8 = {0, 0, 0, 0, 0, 0, 0, 0};
    __builtin_amdgcn_tensor_load_to_lds(g0, g1, z, z, z8, 0);
#else
    __builtin_amdgcn_tensor_load_to_lds(g0, g1, z, z, 0);
#endif
}
#endif

// =====================================================================
// Guard-free bf16 GEMM:  C[M,N] = epi( A[M,K](lda,bf16) * W[N,K](bf16)^T )
// M%128==0, N%64==0, K%32==0.  fp32 accumulation via v_wmma_f32_16x16x32_bf16.
// TDM path: double-buffered LDS, wave 0 issues TENSOR_LOAD_TO_LDS for the
// next k-slab (A+W), s_wait_tensorcnt keeps one slab in flight; DMA overlaps
// the WMMA sequence.  Fallback: register-pipelined b128 copies.
// epi: 0 = plain fp32 store; 1 = softplus(x + bias[n]); 2 = fp32 + bf16 store.
// =====================================================================
#define GBM 128
#define GBN 64
#define GBK 32
#define LDS_STRIDE 40   // bf16 elems: 32 data + 8 pad (= TDM 16+4 dwords)

__global__ __launch_bounds__(256)
void gemm_bf16_wmma(const unsigned short* __restrict__ A,
                    const unsigned short* __restrict__ W,
                    const float* __restrict__ bias,
                    float* __restrict__ C,
                    unsigned short* __restrict__ Cbf,
                    int M, int N, int K, int lda, int epi) {
    const int t    = threadIdx.x;
    const int m0   = blockIdx.y * GBM;
    const int n0   = blockIdx.x * GBN;
    const int lane = t & 31;
    const int wave = t >> 5;
    const int wm   = wave & 3;     // 4 waves along M
    const int wn   = wave >> 2;    // 2 waves along N
    const int lr   = lane & 15;    // row (A) / col (B) within a 16x16 tile
    const int hi   = lane >> 4;    // lane half
    const int kb   = hi * 8;       // K chunks {kb..kb+7, kb+16..kb+23}

    union Frag { uint4 u[2]; v16bf v; };
    v8f acc[2][2] = {};
    const int nslab = K / GBK;

#if USE_TDM
    __shared__ __align__(16) unsigned short sA[2][GBM * LDS_STRIDE]; // 2 x 10 KB
    __shared__ __align__(16) unsigned short sW[2][GBN * LDS_STRIDE]; // 2 x  5 KB
    const unsigned ldsA[2] = { (unsigned)(uintptr_t)&sA[0][0],
                               (unsigned)(uintptr_t)&sA[1][0] };
    const unsigned ldsW[2] = { (unsigned)(uintptr_t)&sW[0][0],
                               (unsigned)(uintptr_t)&sW[1][0] };
    const unsigned short* abase = A + (size_t)m0 * lda;
    const unsigned short* wbase = W + (size_t)n0 * K;

    if (wave == 0) {   // prologue: slab 0 -> buffer 0
        tdm_load_tile_2d(ldsA[0], abase, lda, M, lda, GBK, GBM);
        tdm_load_tile_2d(ldsW[0], wbase, K,   N, K,   GBK, GBN);
    }
    for (int i = 0; i < nslab; ++i) {
        const int cur = i & 1, nxt = cur ^ 1;
        __syncthreads();                       // buf[nxt] free for refill
        if (wave == 0) {
            if (i + 1 < nslab) {
                tdm_load_tile_2d(ldsA[nxt], abase + (i + 1) * GBK, lda, M, lda, GBK, GBM);
                tdm_load_tile_2d(ldsW[nxt], wbase + (i + 1) * GBK, K,   N, K,   GBK, GBN);
                __builtin_amdgcn_s_wait_tensorcnt(2);   // slab i landed
            } else {
                __builtin_amdgcn_s_wait_tensorcnt(0);
            }
        }
        __syncthreads();
        const unsigned short* sAc = &sA[cur][0];
        const unsigned short* sWc = &sW[cur][0];
#else
    __shared__ __align__(16) unsigned short sA1[GBM * LDS_STRIDE];
    __shared__ __align__(16) unsigned short sW1[GBN * LDS_STRIDE];
    const int ar = t >> 1, ac = (t & 1) * 16;     // A: 2 thr/row x 16 bf16 (32B)
    const int wr = t >> 2, wc2 = (t & 3) * 8;     // W: 4 thr/row x  8 bf16 (16B)
    const unsigned short* aptr = A + (size_t)(m0 + ar) * lda + ac;
    const unsigned short* wptr = W + (size_t)(n0 + wr) * K + wc2;
    uint4 ra0 = *(const uint4*)(aptr);
    uint4 ra1 = *(const uint4*)(aptr + 8);
    uint4 rw0 = *(const uint4*)(wptr);
    for (int i = 0; i < nslab; ++i) {
        __syncthreads();
        *(uint4*)(sA1 + ar * LDS_STRIDE + ac)     = ra0;
        *(uint4*)(sA1 + ar * LDS_STRIDE + ac + 8) = ra1;
        *(uint4*)(sW1 + wr * LDS_STRIDE + wc2)    = rw0;
        __syncthreads();
        if (i + 1 < nslab) {                   // prefetch next slab
            ra0 = *(const uint4*)(aptr + (i + 1) * GBK);
            ra1 = *(const uint4*)(aptr + (i + 1) * GBK + 8);
            rw0 = *(const uint4*)(wptr + (i + 1) * GBK);
        }
        const unsigned short* sAc = sA1;
        const unsigned short* sWc = sW1;
#endif
        // ---- fragments per documented CDNA5 bf16 layouts ----
        Frag af[2], bf2[2];
        #pragma unroll
        for (int mt = 0; mt < 2; ++mt) {
            const unsigned short* p = sAc + (wm * 32 + mt * 16 + lr) * LDS_STRIDE + kb;
            af[mt].u[0] = *(const uint4*)(p);
            af[mt].u[1] = *(const uint4*)(p + 16);
        }
        #pragma unroll
        for (int nt = 0; nt < 2; ++nt) {
            const unsigned short* p = sWc + (wn * 32 + nt * 16 + lr) * LDS_STRIDE + kb;
            bf2[nt].u[0] = *(const uint4*)(p);
            bf2[nt].u[1] = *(const uint4*)(p + 16);
        }
        #pragma unroll
        for (int mt = 0; mt < 2; ++mt)
            #pragma unroll
            for (int nt = 0; nt < 2; ++nt)
                acc[mt][nt] = __builtin_amdgcn_wmma_f32_16x16x32_bf16(
                    false, af[mt].v, false, bf2[nt].v,
                    (short)0, acc[mt][nt], false, false);
    }

    // ---- epilogue: C/D layout N = lane%16, M = r + 8*(lane/16) ----
    #pragma unroll
    for (int mt = 0; mt < 2; ++mt) {
        #pragma unroll
        for (int nt = 0; nt < 2; ++nt) {
            const int nn = n0 + wn * 32 + nt * 16 + lr;
            const int mbase = m0 + wm * 32 + mt * 16 + 8 * hi;
            #pragma unroll
            for (int r = 0; r < 8; ++r) {
                float v = acc[mt][nt][r];
                if (epi == 1) {
                    float x = v + bias[nn];
                    v = (x > 20.f) ? x : log1pf(expf(x));     // softplus
                }
                C[(size_t)(mbase + r) * N + nn] = v;
                if (epi == 2)
                    Cbf[(size_t)(mbase + r) * N + nn] = (unsigned short)f2bf1(v);
            }
        }
    }
}

// =====================================================================
// Causal depthwise conv (D_CONV=4) along time + bias + SiLU.
// Emits fp32 (for scan) and bf16 (for x_proj GEMM).
// =====================================================================
__global__ __launch_bounds__(256)
void dwconv_silu_kernel(const float* __restrict__ xz, const float* __restrict__ cw,
                        const float* __restrict__ cb, float* __restrict__ u,
                        unsigned short* __restrict__ u_bf) {
    const long long i = (long long)blockIdx.x * blockDim.x + threadIdx.x;
    if (i >= (long long)MROWS * EE) return;
    const int m = (int)(i / EE), e = (int)(i % EE);
    const int b = m / LL, tt = m % LL;

    float acc = cb[e];
    #pragma unroll
    for (int k = 0; k < DCONV; ++k) {
        const int ts = tt - (DCONV - 1) + k;
        if (ts >= 0)
            acc += cw[e * DCONV + k] * xz[(size_t)(b * LL + ts) * (2 * EE) + e];
    }
    const float s = acc / (1.f + expf(-acc));               // silu
    u[i]    = s;
    u_bf[i] = (unsigned short)f2bf1(s);
}

// =====================================================================
// Selective scan: one thread per (b,e) recurrence, 16-wide state in
// registers, fused +u*D and silu(z) gating; emits bf16 y for out_proj.
// =====================================================================
__global__ __launch_bounds__(256)
void scan_kernel(const float* __restrict__ dt, const float* __restrict__ xdb,
                 const float* __restrict__ u, const float* __restrict__ xz,
                 const float* __restrict__ A_log, const float* __restrict__ Dp,
                 unsigned short* __restrict__ y_bf) {
    const int id = blockIdx.x * blockDim.x + threadIdx.x;
    if (id >= BB * EE) return;
    const int b = id / EE, e = id % EE;

    float hs[DSTATE];
    float Av[DSTATE];
    #pragma unroll
    for (int n = 0; n < DSTATE; ++n) {
        hs[n] = 0.f;
        Av[n] = -expf(A_log[(size_t)e * DSTATE + n]);
    }
    const float De = Dp[e];

    for (int t = 0; t < LL; ++t) {
        const size_t m = (size_t)(b * LL + t);
        const float dtv = dt[m * EE + e];
        const float uv  = u[m * EE + e];
        const float du  = dtv * uv;
        const float* Bm = xdb + m * XDB_PAD + DTRANK;
        const float* Cm = Bm + DSTATE;
        float dot = 0.f;
        #pragma unroll
        for (int n = 0; n < DSTATE; ++n) {
            const float dA = expf(dtv * Av[n]);
            hs[n] = dA * hs[n] + du * Bm[n];
            dot  += hs[n] * Cm[n];
        }
        const float zv = xz[m * (2 * EE) + EE + e];
        const float gate = zv / (1.f + expf(-zv));          // silu(z)
        y_bf[m * EE + e] = (unsigned short)f2bf1((dot + uv * De) * gate);
    }
}

// =====================================================================
extern "C" void kernel_launch(void* const* d_in, const int* in_sizes, int n_in,
                              void* d_out, int out_size, void* d_ws, size_t ws_size,
                              hipStream_t stream) {
    (void)in_sizes; (void)n_in; (void)out_size; (void)ws_size;

    const float* x_in   = (const float*)d_in[0];   // (B,L,D)
    const float* ipw    = (const float*)d_in[1];   // (4, 2E, D)
    const float* cw     = (const float*)d_in[2];   // (4, E, 4)
    const float* cbv    = (const float*)d_in[3];   // (4, E)
    const float* xpw    = (const float*)d_in[4];   // (4, 96, E)
    const float* dtw    = (const float*)d_in[5];   // (4, E, 64)
    const float* dtbias = (const float*)d_in[6];   // (4, E)
    const float* alog   = (const float*)d_in[7];   // (4, E, 16)
    const float* dpar   = (const float*)d_in[8];   // (4, E)
    const float* opw    = (const float*)d_in[9];   // (4, D, E)
    const float* nw     = (const float*)d_in[10];  // (4, D)
    const float* nfw    = (const float*)d_in[11];  // (D)
    float* out = (float*)d_out;

    // ---- workspace layout: fp32 region then bf16 region ----
    float* ws    = (float*)d_ws;
    float* resid = ws;                                  // M*D
    float* xz    = resid + (size_t)MROWS * DD;          // M*2E
    float* ubuf  = xz    + (size_t)MROWS * 2 * EE;      // M*E
    float* xdbf  = ubuf  + (size_t)MROWS * EE;          // M*128
    float* dtbuf = xdbf  + (size_t)MROWS * XDB_PAD;     // M*E
    float* hbuf  = dtbuf + (size_t)MROWS * EE;          // M*D
    unsigned short* hn_bf  = (unsigned short*)(hbuf + (size_t)MROWS * DD);
    unsigned short* u_bf   = hn_bf  + (size_t)MROWS * DD;
    unsigned short* xdb_bf = u_bf   + (size_t)MROWS * EE;
    unsigned short* y_bf   = xdb_bf + (size_t)MROWS * XDB_PAD;
    unsigned short* ipw_bf = y_bf   + (size_t)MROWS * EE;
    unsigned short* xpw_bf = ipw_bf + (size_t)NLAYER * 2 * EE * DD;
    unsigned short* dtw_bf = xpw_bf + (size_t)NLAYER * XDB_PAD * EE;
    unsigned short* opw_bf = dtw_bf + (size_t)NLAYER * EE * DTRANK;

    const dim3 blk(256);
    const dim3 g_inproj((2 * EE) / GBN, MROWS / GBM);
    const dim3 g_xproj (XDB_PAD / GBN,  MROWS / GBM);
    const dim3 g_dt    (EE / GBN,       MROWS / GBM);
    const dim3 g_out   (DD / GBN,       MROWS / GBM);
    const dim3 g_conv((unsigned)(((long long)MROWS * EE + 255) / 256));
    const dim3 g_scan((BB * EE + 255) / 256);

    // ---- one-time weight conversion to bf16 ----
    {
        const long long n_ipw = (long long)NLAYER * 2 * EE * DD / 4;
        const long long n_dtw = (long long)NLAYER * EE * DTRANK / 4;
        const long long n_opw = (long long)NLAYER * DD * EE / 4;
        cvt_bf16_kernel<<<(unsigned)((n_ipw + 255) / 256), blk, 0, stream>>>(ipw, ipw_bf, n_ipw);
        cvt_bf16_kernel<<<(unsigned)((n_dtw + 255) / 256), blk, 0, stream>>>(dtw, dtw_bf, n_dtw);
        cvt_bf16_kernel<<<(unsigned)((n_opw + 255) / 256), blk, 0, stream>>>(opw, opw_bf, n_opw);
        pad_xpw_bf16_kernel<<<(NLAYER * XDB_PAD * EE + 255) / 256, blk, 0, stream>>>(xpw, xpw_bf);
    }

    for (int layer = 0; layer < NLAYER; ++layer) {
        const float* hcur = (layer == 0) ? x_in : hbuf;

        // residual add + prenorm -> bf16 feed for in_proj
        addnorm_kernel<<<MROWS, blk, 0, stream>>>(
            hcur, resid, resid, nw + (size_t)layer * DD, nullptr, hn_bf, layer > 0);

        // in_proj: xz[M,2E] = hn[M,D] * ipw^T
        gemm_bf16_wmma<<<g_inproj, blk, 0, stream>>>(
            hn_bf, ipw_bf + (size_t)layer * 2 * EE * DD, nullptr, xz, nullptr,
            MROWS, 2 * EE, DD, DD, 0);

        // depthwise conv + silu -> fp32 + bf16
        dwconv_silu_kernel<<<g_conv, blk, 0, stream>>>(
            xz, cw + (size_t)layer * EE * DCONV, cbv + (size_t)layer * EE, ubuf, u_bf);

        // x_proj (padded): xdb[M,128] = u[M,E] * xpwpad^T  (fp32 + bf16 out)
        gemm_bf16_wmma<<<g_xproj, blk, 0, stream>>>(
            u_bf, xpw_bf + (size_t)layer * XDB_PAD * EE, nullptr, xdbf, xdb_bf,
            MROWS, XDB_PAD, EE, EE, 2);

        // dt_proj + softplus: dt[M,E] = softplus(xdb[:, :64] * dtw^T + dtb)
        gemm_bf16_wmma<<<g_dt, blk, 0, stream>>>(
            xdb_bf, dtw_bf + (size_t)layer * EE * DTRANK, dtbias + (size_t)layer * EE,
            dtbuf, nullptr, MROWS, EE, DTRANK, XDB_PAD, 1);

        // selective scan (+ u*D, * silu(z)) -> bf16 y
        scan_kernel<<<g_scan, blk, 0, stream>>>(
            dtbuf, xdbf, ubuf, xz,
            alog + (size_t)layer * EE * DSTATE, dpar + (size_t)layer * EE, y_bf);

        // out_proj: h[M,D] = y[M,E] * opw^T
        gemm_bf16_wmma<<<g_out, blk, 0, stream>>>(
            y_bf, opw_bf + (size_t)layer * DD * EE, nullptr, hbuf, nullptr,
            MROWS, DD, EE, EE, 0);
    }

    // final fused add + RMSNorm -> d_out (fp32)
    addnorm_kernel<<<MROWS, blk, 0, stream>>>(hbuf, resid, resid, nfw, out, nullptr, 1);
}
